// GCN_83434034692317
// MI455X (gfx1250) — matrix operand order repaired and checked by reading, not compiled, not run on hardware
//
#include <hip/hip_runtime.h>

// ---------------------------------------------------------------------------
// GCN on MI455X (gfx1250): bf16 WMMA GEMMs (32x64 tile/wave) +
// L2-resident atomic aggregation.
// ---------------------------------------------------------------------------

typedef __bf16 bf16_t;
typedef __attribute__((ext_vector_type(16))) __bf16 v16bf;
typedef __attribute__((ext_vector_type(8)))  __bf16 v8bf;
typedef __attribute__((ext_vector_type(8)))  float  v8f;

static __device__ __forceinline__ v16bf cat8(v8bf lo, v8bf hi)
{
    return __builtin_shufflevector(lo, hi,
                0, 1, 2, 3, 4, 5, 6, 7, 8, 9, 10, 11, 12, 13, 14, 15);
}

// ---------------------------------------------------------------------------
// Repack W (f32, K x Nd row-major) into WMMA B-operand tiles (bf16).
// Tile (kt, nt): 32 lanes x 16 bf16 contiguous.
//   lane l: n = l & 15, k0 = (l >= 16) ? 8 : 0
//   j-th value = W[kt*32 + k0 + j + (j >= 8 ? 8 : 0)][nt*16 + n]   (j = 0..15)
// Matches V_WMMA_*_16X16X32 B layout (ISA 7.12.2): VGPR v, lanes 0-15 hold
// K=2v,2v+1; lanes 16-31 hold K=8+2v,8+2v+1 (v>=4: both +8).
// ---------------------------------------------------------------------------
__global__ void gcn2_repack_w(const float* __restrict__ W, bf16_t* __restrict__ out,
                              int K, int Nd)
{
    int t = blockIdx.x * blockDim.x + threadIdx.x;
    int total = (K >> 5) * (Nd >> 4) * 32;
    if (t >= total) return;
    int lane = t & 31;
    int tile = t >> 5;
    int ntn  = Nd >> 4;
    int nt   = tile % ntn;
    int kt   = tile / ntn;
    int n    = lane & 15;
    int k0   = (lane >> 4) << 3;            // 0 or 8
    bf16_t* dst = out + (size_t)t * 16;
#pragma unroll
    for (int j = 0; j < 16; ++j) {
        float v = W[(size_t)(kt * 32 + k0 + j + (j >= 8 ? 8 : 0)) * Nd + nt * 16 + n];
        dst[j] = (bf16_t)v;
    }
}

// ---------------------------------------------------------------------------
// GEMM: Out[M x Nd] = A[M x K] @ Bpk (+bias). bf16 in, f32 accumulate/out.
// One wave computes a 32x64 output tile: 2 A-frags x 4 B-frags -> 8 WMMAs
// per 32-wide K-step (A reused 4x, B reused 2x).
// A operand layout (ISA 7.12.2, 16-bit A 16x32):
//   lane l: m = l & 15, ka = (l >= 16) ? 8 : 0
//   first 8 bf16 = A[m][kbase+ka .. +7], next 8 = A[m][kbase+16+ka .. +7]
// ---------------------------------------------------------------------------
__global__ __launch_bounds__(128)
void gcn2_gemm_bf16_wmma(const bf16_t* __restrict__ Abuf,   // Mpad x K row-major
                         const bf16_t* __restrict__ Bpk,    // packed tiles
                         float* __restrict__ Out,           // row-major, stride Nd
                         const float* __restrict__ bias,    // may be nullptr
                         int K, int Nd, int Nstore)
{
    const int lane = threadIdx.x & 31;
    const int wave = threadIdx.x >> 5;
    const int mt0  = blockIdx.x * 2;               // two 16-row subtiles
    const int nt0  = (blockIdx.y * 4 + wave) * 4;  // four 16-col subtiles
    const int m    = lane & 15;
    const int g    = lane >> 4;                    // lane half
    const int ktiles = K >> 5;
    const int ntn    = Nd >> 4;

    const bf16_t* ap0 = Abuf + (size_t)(mt0 * 16 + m) * K + (g << 3);
    const bf16_t* ap1 = ap0 + (size_t)16 * K;
    const bf16_t* bp  = Bpk + ((size_t)nt0 * 32 + lane) * 16;
    const size_t  bstride = (size_t)ntn * 32 * 16;  // elements per K-step

    v8f acc[2][4];
#pragma unroll
    for (int i = 0; i < 2; ++i)
#pragma unroll
        for (int j = 0; j < 4; ++j) acc[i][j] = (v8f){};

    for (int kt = 0; kt < ktiles; ++kt) {
        const bf16_t* pa0 = ap0 + kt * 32;
        const bf16_t* pa1 = ap1 + kt * 32;
        v16bf a0 = cat8(*(const v8bf*)pa0, *(const v8bf*)(pa0 + 16));
        v16bf a1 = cat8(*(const v8bf*)pa1, *(const v8bf*)(pa1 + 16));
        const bf16_t* bkt = bp + (size_t)kt * bstride;
#pragma unroll
        for (int j = 0; j < 4; ++j) {
            v16bf b = *(const v16bf*)(bkt + j * 512);   // subtiles 1KB apart
            acc[0][j] = __builtin_amdgcn_wmma_f32_16x16x32_bf16(
                            false, a0, false, b, (short)0, acc[0][j], false, false);
            acc[1][j] = __builtin_amdgcn_wmma_f32_16x16x32_bf16(
                            false, a1, false, b, (short)0, acc[1][j], false, false);
        }
    }

    // C/D layout: VGPR r -> row (mt0+i)*16 + g*8 + r, col (nt0+j)*16 + (lane&15)
#pragma unroll
    for (int i = 0; i < 2; ++i) {
        const int mb = (mt0 + i) * 16 + (g << 3);
#pragma unroll
        for (int j = 0; j < 4; ++j) {
            const int col = (nt0 + j) * 16 + (lane & 15);
            const float bv = bias ? bias[col] : 0.0f;
#pragma unroll
            for (int r = 0; r < 8; ++r) {
                int orow = mb + r;
                if (orow < Nstore) Out[(size_t)orow * Nd + col] = acc[i][j][r] + bv;
            }
        }
    }
}

// ---------------------------------------------------------------------------
// Edge-degree + normalization
// ---------------------------------------------------------------------------
__global__ void gcn2_degree(const int* __restrict__ dst, float* __restrict__ deg, int E)
{
    int e = blockIdx.x * blockDim.x + threadIdx.x;
    if (e < E) atomicAdd(&deg[dst[e]], 1.0f);
}

__global__ void gcn2_rsqrt_deg(float* __restrict__ deg, int n)
{
    int i = blockIdx.x * blockDim.x + threadIdx.x;
    if (i < n) deg[i] = rsqrtf(deg[i] + 1.0f);   // +1 self-loop
}

// ---------------------------------------------------------------------------
// Edge aggregation: B[dst] += A[src] * dinv[src]*dinv[dst]. One wave per edge,
// float4 reads + scalar f32 atomics (no-return -> STOREcnt path).
// ---------------------------------------------------------------------------
__global__ __launch_bounds__(256)
void gcn2_aggregate(const float* __restrict__ A, float* __restrict__ B,
                    const int* __restrict__ src, const int* __restrict__ dst,
                    const float* __restrict__ dinv, int E, int Nd)
{
    int e = blockIdx.x * 8 + (threadIdx.x >> 5);
    if (e >= E) return;
    int lane = threadIdx.x & 31;
    int s = src[e], d = dst[e];
    float w = dinv[s] * dinv[d];
    const float4* ap = (const float4*)(A + (size_t)s * Nd);
    float* bp = B + (size_t)d * Nd;
    for (int c4 = lane; c4 < (Nd >> 2); c4 += 32) {
        float4 v = ap[c4];
        int c = c4 << 2;
        atomicAdd(&bp[c + 0], v.x * w);
        atomicAdd(&bp[c + 1], v.y * w);
        atomicAdd(&bp[c + 2], v.z * w);
        atomicAdd(&bp[c + 3], v.w * w);
    }
}

// ---------------------------------------------------------------------------
// Self-loop term + bias + leaky_relu (slope==1.0 -> identity).
// ---------------------------------------------------------------------------
__global__ void gcn2_finish(float* __restrict__ B, const float* __restrict__ A,
                            const float* __restrict__ dinv, const float* __restrict__ bias,
                            float slope, int Nd)
{
    int i = blockIdx.x;
    float w = dinv[i] * dinv[i];
    for (int c = threadIdx.x; c < Nd; c += blockDim.x) {
        size_t idx = (size_t)i * Nd + c;
        float v = B[idx] + A[idx] * w + bias[c];
        B[idx] = (v >= 0.0f) ? v : slope * v;
    }
}

// ---------------------------------------------------------------------------
// BatchNorm stats: per-column sum / sum-of-squares. Column-per-lane,
// grid-stride over rows, one atomic per block per column.
// ---------------------------------------------------------------------------
__global__ __launch_bounds__(512)
void gcn2_bnstats(const float* __restrict__ B, float* __restrict__ colsum,
                  float* __restrict__ colsq, int N, int Nd)
{
    int c = threadIdx.x;                          // blockDim.x == Nd == 512
    float s = 0.0f, q = 0.0f;
    for (int i = blockIdx.x; i < N; i += gridDim.x) {
        float v = B[(size_t)i * Nd + c];
        s += v; q += v * v;
    }
    atomicAdd(&colsum[c], s);
    atomicAdd(&colsq[c], q);
}

// ---------------------------------------------------------------------------
// BN apply (biased var) fused with f32 -> bf16 conversion for next GEMM.
// ---------------------------------------------------------------------------
__global__ void gcn2_bnapply(const float* __restrict__ B, bf16_t* __restrict__ Hbf,
                             const float* __restrict__ colsum, const float* __restrict__ colsq,
                             const float* __restrict__ gmm, const float* __restrict__ beta,
                             int N, int Nd)
{
    int i = blockIdx.x;
    float invN = 1.0f / (float)N;
    for (int c = threadIdx.x; c < Nd; c += blockDim.x) {
        float mu  = colsum[c] * invN;
        float var = colsq[c] * invN - mu * mu;
        float sc  = gmm[c] * rsqrtf(var + 1e-5f);
        float v   = (B[(size_t)i * Nd + c] - mu) * sc + beta[c];
        Hbf[(size_t)i * Nd + c] = (bf16_t)v;
    }
}

__global__ void gcn2_cvt_bf16(const float* __restrict__ X, bf16_t* __restrict__ Xbf, int Nd)
{
    int i = blockIdx.x;
    for (int c = threadIdx.x; c < Nd; c += blockDim.x)
        Xbf[(size_t)i * Nd + c] = (bf16_t)X[(size_t)i * Nd + c];
}

// ---------------------------------------------------------------------------
// Host orchestration (graph-capture safe: kernels + hipMemsetAsync only).
// ---------------------------------------------------------------------------
extern "C" void kernel_launch(void* const* d_in, const int* in_sizes, int n_in,
                              void* d_out, int out_size, void* d_ws, size_t ws_size,
                              hipStream_t stream)
{
    (void)n_in; (void)out_size; (void)ws_size;
    const int DIN = 256, DH = 512;
    const int N    = in_sizes[0] / DIN;
    const int E    = in_sizes[1] / 2;
    const int MPAD = (N + 31) & ~31;          // 32-row GEMM tiles, EXEC all-1s

    const float* x    = (const float*)d_in[0];
    const int*   ei   = (const int*)d_in[1];
    const int*   srcI = ei;
    const int*   dstI = ei + E;

    const float *W[6], *bvec[6], *gvec[5], *betav[5];
    for (int i = 0; i < 5; ++i) {
        W[i]     = (const float*)d_in[2 + 4 * i];
        bvec[i]  = (const float*)d_in[3 + 4 * i];
        gvec[i]  = (const float*)d_in[4 + 4 * i];
        betav[i] = (const float*)d_in[5 + 4 * i];
    }
    W[5]    = (const float*)d_in[22];
    bvec[5] = (const float*)d_in[23];

    // ---- workspace carve-up (256B-aligned) ----
    char* ws = (char*)d_ws;
    size_t off = 0;
    auto take = [&](size_t bytes) -> char* {
        char* p = ws + off;
        off = (off + bytes + 255) & ~(size_t)255;
        return p;
    };
    bf16_t* xbf   = (bf16_t*)take((size_t)MPAD * DIN * sizeof(bf16_t));
    bf16_t* hbf   = (bf16_t*)take((size_t)MPAD * DH * sizeof(bf16_t));
    float*  Abuf  = (float*) take((size_t)MPAD * DH * sizeof(float));
    float*  Bbuf  = (float*) take((size_t)N * DH * sizeof(float));
    float*  dinv  = (float*) take((size_t)N * sizeof(float));
    float*  stats = (float*) take((size_t)2 * DH * sizeof(float));
    bf16_t* wpk   = (bf16_t*)take((size_t)(DIN * DH + 5 * DH * DH) * sizeof(bf16_t));

    // ---- degree / normalization ----
    hipMemsetAsync(dinv, 0, (size_t)N * sizeof(float), stream);
    gcn2_degree<<<dim3((E + 255) / 256), 256, 0, stream>>>(dstI, dinv, E);
    gcn2_rsqrt_deg<<<dim3((N + 255) / 256), 256, 0, stream>>>(dinv, N);

    // ---- convert x, repack all weights into WMMA B layout ----
    gcn2_cvt_bf16<<<dim3(N), 256, 0, stream>>>(x, xbf, DIN);

    size_t woff[6];
    woff[0] = 0;
    for (int i = 1; i < 5; ++i) woff[i] = (size_t)DIN * DH + (size_t)(i - 1) * DH * DH;
    woff[5] = (size_t)DIN * DH + (size_t)4 * DH * DH;
    for (int i = 0; i < 6; ++i) {
        int K = (i == 0) ? DIN : DH;
        int threads = (K >> 5) * (DH >> 4) * 32;
        gcn2_repack_w<<<dim3((threads + 255) / 256), 256, 0, stream>>>(W[i], wpk + woff[i], K, DH);
    }

    // grid: x = 32-row tiles, y = 256-col groups (4 waves x 64 cols)
    const dim3 gemm_grid(MPAD / 32, DH / 256);

    // ---- 5 conv layers ----
    for (int layer = 0; layer < 5; ++layer) {
        const bf16_t* inb = (layer == 0) ? xbf : hbf;
        const int K = (layer == 0) ? DIN : DH;

        gcn2_gemm_bf16_wmma<<<gemm_grid, 128, 0, stream>>>(
            inb, wpk + woff[layer], Abuf, nullptr, K, DH, MPAD);

        hipMemsetAsync(Bbuf, 0, (size_t)N * DH * sizeof(float), stream);
        gcn2_aggregate<<<dim3((E + 7) / 8), 256, 0, stream>>>(Abuf, Bbuf, srcI, dstI, dinv, E, DH);
        gcn2_finish<<<dim3(N), 256, 0, stream>>>(Bbuf, Abuf, dinv, bvec[layer],
                                                 (layer < 4) ? 0.2f : 1.0f, DH);

        hipMemsetAsync(stats, 0, (size_t)2 * DH * sizeof(float), stream);
        gcn2_bnstats<<<dim3(104), 512, 0, stream>>>(Bbuf, stats, stats + DH, N, DH);
        gcn2_bnapply<<<dim3(N), 256, 0, stream>>>(Bbuf, hbf, stats, stats + DH,
                                                  gvec[layer], betav[layer], N, DH);
    }

    // ---- final linear: d_out = h @ Wf + bf ----
    gcn2_gemm_bf16_wmma<<<gemm_grid, 128, 0, stream>>>(
        hbf, wpk + woff[5], (float*)d_out, bvec[5], DH, DH, N);
}
